// NTXentLoss_26809185861696
// MI455X (gfx1250) — matrix operand order repaired and checked by reading, not compiled, not run on hardware
//
#include <hip/hip_runtime.h>
#include <math.h>

typedef __attribute__((ext_vector_type(16))) _Float16 v16h;
typedef __attribute__((ext_vector_type(8)))  float    v8f;

#define B_ROWS 4096
#define N_ROWS 8192
#define DIMS   256
#define NTILES (N_ROWS / 16)
#define INV_T  14.285714285714286f   // 1 / 0.07

// ---------------------------------------------------------------------------
// CDNA5 async global->LDS DMA (ASYNCcnt-tracked), per CDNA5 ISA
// the LDS destination is the LDS base plus the lane's VDST offset; flat shared addrs carry the LDS offset in
// the low 32 bits (aperture in the high 32), so truncating a generic shared
// pointer yields the VDST value.
// ---------------------------------------------------------------------------
__device__ __forceinline__ void async_ld16(const _Float16* lds_dst, const _Float16* gsrc) {
    asm volatile("global_load_async_to_lds_b128 %0, %1, off"
                 :: "v"((uint32_t)(uintptr_t)lds_dst),
                    "v"((unsigned long long)(uintptr_t)gsrc)
                 : "memory");
}
__device__ __forceinline__ void wait_async0() {
    asm volatile("s_wait_asynccnt 0" ::: "memory");
}

// ---------------------------------------------------------------------------
// Kernel 1: L2-normalize rows of [z_i; z_j] and pack to f16 in workspace.
// ---------------------------------------------------------------------------
__global__ void __launch_bounds__(256) ntxent_normalize(const float* __restrict__ zi,
                                                        const float* __restrict__ zj,
                                                        _Float16* __restrict__ feats) {
    const int row = blockIdx.x;
    const int tid = threadIdx.x;
    const float* src = (row < B_ROWS) ? (zi + (size_t)row * DIMS)
                                      : (zj + (size_t)(row - B_ROWS) * DIMS);
    float x = src[tid];
    float ss = x * x;
    #pragma unroll
    for (int m = 16; m >= 1; m >>= 1) ss += __shfl_xor(ss, m, 32);

    __shared__ float wss[8];
    if ((tid & 31) == 0) wss[tid >> 5] = ss;
    __syncthreads();
    float tot = 0.f;
    #pragma unroll
    for (int w = 0; w < 8; ++w) tot += wss[w];

    const float inv = 1.0f / sqrtf(tot);
    feats[(size_t)row * DIMS + tid] = (_Float16)(x * inv);
}

// ---------------------------------------------------------------------------
// Kernel 2: streaming sim + logsumexp.  256 threads = 8 waves per block; wave
// w owns 16-row tile (blockIdx.x*8 + w).  Column tiles (16 x 256 f16, 8 KB)
// are double-buffered in LDS via async global->LDS DMA: loads for tile nt+1
// are issued before computing tile nt, hiding the L2 latency under the WMMAs.
// Fixed logsumexp max = 1/T (valid since |sim| <= 1 after normalization).
// ---------------------------------------------------------------------------
__global__ void __launch_bounds__(256, 1) ntxent_main(const _Float16* __restrict__ feats,
                                                      float* __restrict__ rowloss) {
    __shared__ _Float16 tileB[2][16 * DIMS];   // 2 x 8 KB double buffer

    const int tid   = threadIdx.x;
    const int lane  = tid & 31;
    const int wave  = tid >> 5;
    const int half  = lane >> 4;     // 0: lanes 0-15, 1: lanes 16-31
    const int l16   = lane & 15;
    const int m_base = (blockIdx.x * 8 + wave) * 16;

    // ---- Preload A tile (16 rows x 256 dims) into 8 register fragments -----
    // ISA 16-bit A 16x32 layout: halves 0..7 = K kc*32 + half*8 + {0..7},
    //                            halves 8..15 = K kc*32 + 16 + half*8 + {0..7}
    const _Float16* arp = feats + (size_t)(m_base + l16) * DIMS;
    v16h A[8];
    #pragma unroll
    for (int kc = 0; kc < 8; ++kc) {
        const int k0 = kc * 32 + half * 8;
        const int k1 = kc * 32 + 16 + half * 8;
        *(uint4*)&A[kc]         = *(const uint4*)(arp + k0);
        *(((uint4*)&A[kc]) + 1) = *(const uint4*)(arp + k1);
    }

    float s[8], pos[8];
    #pragma unroll
    for (int r = 0; r < 8; ++r) { s[r] = 0.f; pos[r] = 0.f; }

    const int ldr = tid >> 4;   // 0..15: row of column tile this thread stages
    const int ldc = tid & 15;   // 0..15: 16-half chunk within the row

    // Prologue: kick off async DMA of column tile 0 into buffer 0 (32B/thread)
    {
        const _Float16* gp = feats + (size_t)ldr * DIMS + ldc * 16;
        const _Float16* lp = &tileB[0][ldr * DIMS + ldc * 16];
        async_ld16(lp,     gp);
        async_ld16(lp + 8, gp + 8);
    }

    for (int nt = 0; nt < NTILES; ++nt) {
        wait_async0();       // this wave's DMA into buf[nt&1] has landed
        __syncthreads();     // all waves' DMA visible; prior reads of the
                             // other buffer are complete (dscnt waited pre-WMMA)

        // Issue async DMA for the NEXT column tile into the other buffer.
        if (nt + 1 < NTILES) {
            const _Float16* gp = feats + (size_t)((nt + 1) * 16 + ldr) * DIMS + ldc * 16;
            const _Float16* lp = &tileB[(nt + 1) & 1][ldr * DIMS + ldc * 16];
            async_ld16(lp,     gp);
            async_ld16(lp + 8, gp + 8);
        }

        // ---- Load all 8 B fragments, then chain 8 WMMAs back-to-back -------
        // B 32x16 layout: lane holds column l16, K = kc*32 + half*8 + {0..7}, kc*32 + 16 + half*8 + {0..7}
        const _Float16* brp = &tileB[nt & 1][l16 * DIMS + half * 8];
        v16h bf[8];
        #pragma unroll
        for (int kc = 0; kc < 8; ++kc) {
            *(uint4*)&bf[kc]         = *(const uint4*)(brp + kc * 32);
            *(((uint4*)&bf[kc]) + 1) = *(const uint4*)(brp + kc * 32 + 16);
        }
        v8f c = {};
        #pragma unroll
        for (int kc = 0; kc < 8; ++kc)
            c = __builtin_amdgcn_wmma_f32_16x16x32_f16(
                    false, A[kc], false, bf[kc], (short)0, c, false, false);

        // ---- Fold tile into running exp-sums -------------------------------
        // C layout: c[r] is (row m_base + r + 8*half, col nt*16 + l16)
        const int n_glob = nt * 16 + l16;
        #pragma unroll
        for (int r = 0; r < 8; ++r) {
            const int row_glob = m_base + r + half * 8;
            const float logit = c[r] * INV_T;
            if (n_glob == (row_glob ^ B_ROWS)) pos[r] = logit;   // partner view
            const float e = __expf(logit - INV_T);               // <= 1
            s[r] += (n_glob == row_glob) ? 0.f : e;              // mask diagonal
        }
    }

    // ---- Reduce across the 16 lanes of each half (wave32 xor-shuffles) -----
    #pragma unroll
    for (int m = 1; m < 16; m <<= 1) {
        #pragma unroll
        for (int r = 0; r < 8; ++r) {
            s[r]   += __shfl_xor(s[r],   m, 32);
            pos[r] += __shfl_xor(pos[r], m, 32);   // exactly one lane nonzero
        }
    }

    if (l16 == 0) {   // lane 0 writes rows +0..7, lane 16 writes rows +8..15
        #pragma unroll
        for (int r = 0; r < 8; ++r) {
            const int row_glob = m_base + r + half * 8;
            rowloss[row_glob] = (INV_T + __logf(s[r])) - pos[r];
        }
    }
}

// ---------------------------------------------------------------------------
// Kernel 3: deterministic mean of the 8192 per-row losses.
// ---------------------------------------------------------------------------
__global__ void __launch_bounds__(256) ntxent_reduce(const float* __restrict__ rowloss,
                                                     float* __restrict__ out) {
    __shared__ float sm[256];
    float acc = 0.f;
    for (int i = threadIdx.x; i < N_ROWS; i += 256) acc += rowloss[i];
    sm[threadIdx.x] = acc;
    __syncthreads();
    for (int sft = 128; sft > 0; sft >>= 1) {
        if (threadIdx.x < sft) sm[threadIdx.x] += sm[threadIdx.x + sft];
        __syncthreads();
    }
    if (threadIdx.x == 0) out[0] = sm[0] / (float)N_ROWS;
}

// ---------------------------------------------------------------------------
extern "C" void kernel_launch(void* const* d_in, const int* in_sizes, int n_in,
                              void* d_out, int out_size, void* d_ws, size_t ws_size,
                              hipStream_t stream) {
    (void)in_sizes; (void)n_in; (void)out_size; (void)ws_size;

    const float* zi = (const float*)d_in[0];
    const float* zj = (const float*)d_in[1];
    float* out = (float*)d_out;

    // workspace layout: [0, 4MB) f16 normalized feats; then 8192 f32 row losses
    _Float16* feats = (_Float16*)d_ws;
    float* rowloss  = (float*)((char*)d_ws + (size_t)N_ROWS * DIMS * sizeof(_Float16));

    ntxent_normalize<<<N_ROWS, 256, 0, stream>>>(zi, zj, feats);
    ntxent_main<<<(N_ROWS / 16) / 8, 256, 0, stream>>>(feats, rowloss);  // 64 blocks
    ntxent_reduce<<<1, 256, 0, stream>>>(rowloss, out);
}
